// CausalSelfAttention_23330262351975
// MI455X (gfx1250) — matrix operand order repaired and checked by reading, not compiled, not run on hardware
//
#include <hip/hip_runtime.h>
#include <cstdint>
#include <cstddef>

// ---------------- CDNA5 (gfx1250) types ----------------
typedef __attribute__((ext_vector_type(16))) __bf16 v16bf;
typedef __attribute__((ext_vector_type(8)))  float  v8f;

union FragB16 { v16bf v; uint4 u[2]; };   // 32 bytes = 16 bf16 = 8 VGPRs

#define NB 2
#define TT 2048
#define CC 2048
#define HH 16
#define DD 128
#define C3 6144
#define MM 4096   // NB*TT

__device__ __forceinline__ unsigned short f2bf(float f) {
  unsigned int u = __float_as_uint(f);
  u += 0x7FFFu + ((u >> 16) & 1u);        // round-to-nearest-even
  return (unsigned short)(u >> 16);
}
__device__ __forceinline__ float bf2f(unsigned short h) {
  return __uint_as_float(((unsigned int)h) << 16);
}

// ---------------- prep: f32 -> bf16 ----------------
__global__ void k_cvt_bf16(const float* __restrict__ in,
                           unsigned short* __restrict__ out, int n) {
  int i = blockIdx.x * blockDim.x + threadIdx.x;
  int stride = gridDim.x * blockDim.x;
  for (; i < n; i += stride) out[i] = f2bf(in[i]);
}

// in [rows, cols] f32  ->  out [cols, rows] bf16  (so B-fragments are contiguous)
__global__ void k_transpose_bf16(const float* __restrict__ in,
                                 unsigned short* __restrict__ out,
                                 int rows, int cols) {
  int i = blockIdx.x * blockDim.x + threadIdx.x;
  int n = rows * cols;
  int stride = gridDim.x * blockDim.x;
  for (; i < n; i += stride) {
    int r = i / cols, c = i - r * cols;
    out[(size_t)c * rows + r] = f2bf(in[i]);
  }
}

// ---------------- GEMM: out[M,N] = A[M,K] @ Wt[N,K]^T + bias ----------------
// One wave = 32x64 output tile (2 M-tiles x 4 N-tiles), K double-buffered.
__global__ void __launch_bounds__(128)
k_gemm_bf16(const unsigned short* __restrict__ A,
            const unsigned short* __restrict__ Wt,
            const float* __restrict__ bias,
            unsigned short* __restrict__ outBf,
            float* __restrict__ outF,
            int M, int N, int K, int storeF32) {
  const int lane = threadIdx.x & 31;
  const int wid  = threadIdx.x >> 5;
  const int w = blockIdx.x * 4 + wid;
  const int ngroups = N >> 6;
  const int mt = w / ngroups;
  const int ng = w - mt * ngroups;
  if (mt * 32 >= M) return;              // wave-uniform guard
  const int n0 = ng * 64;
  const int m  = lane & 15;
  const int hi = lane >> 4;

  const unsigned short* arow[2];
#pragma unroll
  for (int i = 0; i < 2; ++i)
    arow[i] = A + (size_t)(mt * 32 + i * 16 + m) * K + hi * 8;
  const unsigned short* brow[4];
#pragma unroll
  for (int t = 0; t < 4; ++t)
    brow[t] = Wt + (size_t)(n0 + t * 16 + m) * K + hi * 8;

  v8f acc[8] = {};                        // acc[mi*4 + t]
  const int kc = K >> 5;

  auto loadch = [&](int c, FragB16* af, FragB16* bf) {
#pragma unroll
    for (int i = 0; i < 2; ++i) {
      const unsigned short* p = arow[i] + (size_t)c * 32;
      af[i].u[0] = *(const uint4*)(p);        // K = kb .. kb+7
      af[i].u[1] = *(const uint4*)(p + 16);   // K = kb+16 .. kb+23
    }
#pragma unroll
    for (int t = 0; t < 4; ++t) {
      const unsigned short* p = brow[t] + (size_t)c * 32;
      bf[t].u[0] = *(const uint4*)(p);
      bf[t].u[1] = *(const uint4*)(p + 16);
    }
    __builtin_prefetch(arow[0] + (size_t)(c + 8) * 32, 0, 1);  // global_prefetch
  };

  FragB16 aP[2], bP[4], aQ[2], bQ[4];     // ping-pong buffers
  loadch(0, aP, bP);
  for (int c = 0; c < kc; c += 2) {
    loadch(c + 1, aQ, bQ);                // issue next-chunk loads first
#pragma unroll
    for (int t = 0; t < 4; ++t) {
      acc[0 + t] = __builtin_amdgcn_wmma_f32_16x16x32_bf16(
          false, aP[0].v, false, bP[t].v, (short)0, acc[0 + t], false, false);
      acc[4 + t] = __builtin_amdgcn_wmma_f32_16x16x32_bf16(
          false, aP[1].v, false, bP[t].v, (short)0, acc[4 + t], false, false);
    }
    const int c2 = (c + 2 < kc) ? c + 2 : c;   // clamp (value unused on last iter)
    loadch(c2, aP, bP);
#pragma unroll
    for (int t = 0; t < 4; ++t) {
      acc[0 + t] = __builtin_amdgcn_wmma_f32_16x16x32_bf16(
          false, aQ[0].v, false, bQ[t].v, (short)0, acc[0 + t], false, false);
      acc[4 + t] = __builtin_amdgcn_wmma_f32_16x16x32_bf16(
          false, aQ[1].v, false, bQ[t].v, (short)0, acc[4 + t], false, false);
    }
  }
  // store: C/D layout row = r + 8*hi, col = lane&15 within tile
#pragma unroll
  for (int mi = 0; mi < 2; ++mi) {
#pragma unroll
    for (int t = 0; t < 4; ++t) {
#pragma unroll
      for (int r = 0; r < 8; ++r) {
        int row = mt * 32 + mi * 16 + r + hi * 8;
        int col = n0 + t * 16 + m;
        float v = acc[mi * 4 + t][r] + bias[col];
        if (storeF32) outF[(size_t)row * N + col] = v;
        else          outBf[(size_t)row * N + col] = f2bf(v);
      }
    }
  }
}

// ---------------- RoPE + reorganize: qkv[M,3C] -> Q,K [B,H,T,D], V^T [B,H,D,T] ---
__global__ void k_rope(const unsigned short* __restrict__ qkv,
                       unsigned short* __restrict__ Qr,
                       unsigned short* __restrict__ Kr,
                       unsigned short* __restrict__ Vt) {
  int tid = blockIdx.x * blockDim.x + threadIdx.x;   // 2^22 threads exactly
  int i = tid & 63;
  int t = (tid >> 6) & (TT - 1);
  int h = (tid >> 17) & (HH - 1);
  int b = tid >> 21;
  if (b >= NB) return;
  const size_t base = (size_t)(b * TT + t) * C3 + h * DD;
  float q0 = bf2f(qkv[base + i]);
  float q1 = bf2f(qkv[base + i + 64]);
  float k0 = bf2f(qkv[base + CC + i]);
  float k1 = bf2f(qkv[base + CC + i + 64]);
  float v0 = bf2f(qkv[base + 2 * CC + i]);
  float v1 = bf2f(qkv[base + 2 * CC + i + 64]);
  // inv_freq = 10000^(-i/64) = exp(-i/64 * ln(10000))
  float invf = __expf(-(float)i * (9.210340371976184f / 64.0f));
  float ang = (float)t * invf;
  float sn, cs;
  __sincosf(ang, &sn, &cs);
  float qr0 = q0 * cs - q1 * sn, qr1 = q1 * cs + q0 * sn;
  float kr0 = k0 * cs - k1 * sn, kr1 = k1 * cs + k0 * sn;
  size_t qko = ((size_t)(b * HH + h) * TT + t) * DD;
  Qr[qko + i]      = f2bf(qr0);  Qr[qko + i + 64] = f2bf(qr1);
  Kr[qko + i]      = f2bf(kr0);  Kr[qko + i + 64] = f2bf(kr1);
  size_t vto = (size_t)(b * HH + h) * DD * TT;
  Vt[vto + (size_t)i * TT + t]        = f2bf(v0);
  Vt[vto + (size_t)(i + 64) * TT + t] = f2bf(v1);
}

// ---------------- flash attention: one wave per 16-row Q tile -----------------
__global__ void __launch_bounds__(128)
k_attn(const unsigned short* __restrict__ Q,
       const unsigned short* __restrict__ Kk,
       const unsigned short* __restrict__ Vt,
       unsigned short* __restrict__ Y) {
  __shared__ __align__(16) unsigned short plds[4][16 * 32];  // per-wave P tile
  const int lane = threadIdx.x & 31;
  const int wid  = threadIdx.x >> 5;
  const int w = blockIdx.x * 4 + wid;          // 0..4095
  const int qtile = w & 127;
  const int bh    = w >> 7;                    // b*H + h
  const int qbase = qtile * 16;
  const int n  = lane & 15;
  const int hi = lane >> 4;
  const float scale = 0.08838834764831845f;    // 1/sqrt(128)

  const unsigned short* Qb = Q  + ((size_t)bh * TT + qbase) * DD;
  const unsigned short* Kb = Kk + (size_t)bh * TT * DD;
  const unsigned short* Vb = Vt + (size_t)bh * DD * TT;

  // Q A-fragments for the 4 K-chunks along D=128
  FragB16 qf[4];
#pragma unroll
  for (int c = 0; c < 4; ++c) {
    const unsigned short* p = Qb + (size_t)n * DD + c * 32 + hi * 8;
    qf[c].u[0] = *(const uint4*)(p);
    qf[c].u[1] = *(const uint4*)(p + 16);
  }

  v8f o[8] = {};
  float mrow[8], lrow[8];
#pragma unroll
  for (int r = 0; r < 8; ++r) { mrow[r] = -1e30f; lrow[r] = 0.0f; }

  unsigned short* lds = plds[wid];

  for (int kv0 = 0; kv0 < qbase + 16; kv0 += 32) {
    // ---- batch-load all K fragments for this 32-key chunk (one clause) ----
    FragB16 kf0[4], kf1[4];
#pragma unroll
    for (int c = 0; c < 4; ++c) {
      const unsigned short* p0 = Kb + (size_t)(kv0 + n) * DD + c * 32 + hi * 8;
      const unsigned short* p1 = p0 + (size_t)16 * DD;
      kf0[c].u[0] = *(const uint4*)(p0); kf0[c].u[1] = *(const uint4*)(p0 + 16);
      kf1[c].u[0] = *(const uint4*)(p1); kf1[c].u[1] = *(const uint4*)(p1 + 16);
    }
    // S = Q K^T  (two 16x16 tiles covering 32 keys)
    v8f s0 = {}, s1 = {};
#pragma unroll
    for (int c = 0; c < 4; ++c) {
      s0 = __builtin_amdgcn_wmma_f32_16x16x32_bf16(false, qf[c].v, false, kf0[c].v,
                                                   (short)0, s0, false, false);
      s1 = __builtin_amdgcn_wmma_f32_16x16x32_bf16(false, qf[c].v, false, kf1[c].v,
                                                   (short)0, s1, false, false);
    }
    // ---- batch-load all 8 V fragments (overlaps with softmax math) ----
    FragB16 vf[8];
#pragma unroll
    for (int t = 0; t < 8; ++t) {
      const unsigned short* pv = Vb + (size_t)(t * 16 + n) * TT + kv0 + hi * 8;
      vf[t].u[0] = *(const uint4*)(pv);
      vf[t].u[1] = *(const uint4*)(pv + 16);
    }
    // scale + causal mask (lane-wise cndmask, no divergence)
    const int kc0 = kv0 + n, kc1 = kv0 + 16 + n;
#pragma unroll
    for (int r = 0; r < 8; ++r) {
      const int qrow = qbase + r + hi * 8;
      float a  = s0[r] * scale;
      float bb = s1[r] * scale;
      s0[r] = (kc0 <= qrow) ? a  : -1e30f;
      s1[r] = (kc1 <= qrow) ? bb : -1e30f;
    }
    // online softmax; row r lives in VGPR r across each 16-lane half
    float esc[8];
#pragma unroll
    for (int r = 0; r < 8; ++r) {
      float mx = fmaxf(s0[r], s1[r]);
#pragma unroll
      for (int d = 1; d < 16; d <<= 1) mx = fmaxf(mx, __shfl_xor(mx, d, 16));
      float mnew = fmaxf(mrow[r], mx);
      float sc = __expf(mrow[r] - mnew);
      float p0 = __expf(s0[r] - mnew);
      float p1 = __expf(s1[r] - mnew);
      float ps = p0 + p1;
#pragma unroll
      for (int d = 1; d < 16; d <<= 1) ps += __shfl_xor(ps, d, 16);
      lrow[r] = lrow[r] * sc + ps;
      mrow[r] = mnew;
      esc[r] = sc;
      s0[r] = p0; s1[r] = p1;
    }
#pragma unroll
    for (int t = 0; t < 8; ++t)
#pragma unroll
      for (int r = 0; r < 8; ++r) o[t][r] *= esc[r];

    // P: C/D layout -> LDS -> A-fragment layout (same-wave, DS is in-order)
#pragma unroll
    for (int r = 0; r < 8; ++r) {
      lds[(r + hi * 8) * 32 + n]      = f2bf(s0[r]);
      lds[(r + hi * 8) * 32 + 16 + n] = f2bf(s1[r]);
    }
    asm volatile("s_wait_dscnt 0" ::: "memory");
    FragB16 pf;
    pf.u[0] = *(const uint4*)(lds + n * 32 + hi * 8);
    pf.u[1] = *(const uint4*)(lds + n * 32 + hi * 8 + 16);

    // O += P @ V  (8 d-tiles, V fragments already resident)
#pragma unroll
    for (int t = 0; t < 8; ++t)
      o[t] = __builtin_amdgcn_wmma_f32_16x16x32_bf16(false, pf.v, false, vf[t].v,
                                                     (short)0, o[t], false, false);
  }

  // normalize + store to attn_out [M, C] bf16
  const int b_ = bh >> 4, h_ = bh & 15;
  unsigned short* yb = Y + ((size_t)(b_ * TT) + qbase) * CC + h_ * DD;
#pragma unroll
  for (int r = 0; r < 8; ++r) {
    float inv = 1.0f / lrow[r];
#pragma unroll
    for (int t = 0; t < 8; ++t)
      yb[(size_t)(r + hi * 8) * CC + t * 16 + n] = f2bf(o[t][r] * inv);
  }
}

// ---------------- launch ----------------
extern "C" void kernel_launch(void* const* d_in, const int* in_sizes, int n_in,
                              void* d_out, int out_size, void* d_ws, size_t ws_size,
                              hipStream_t stream) {
  const float* x      = (const float*)d_in[0];
  const float* W_attn = (const float*)d_in[1];
  const float* b_attn = (const float*)d_in[2];
  const float* W_proj = (const float*)d_in[3];
  const float* b_proj = (const float*)d_in[4];
  float* out = (float*)d_out;

  char* ws = (char*)d_ws;                                     // 160 MiB used
  unsigned short* xbf = (unsigned short*)(ws);                // 16 MiB  x bf16
  unsigned short* WtA = (unsigned short*)(ws + (16u  << 20)); // 24 MiB W_attn^T
  unsigned short* WtP = (unsigned short*)(ws + (40u  << 20)); //  8 MiB W_proj^T
  unsigned short* qkv = (unsigned short*)(ws + (48u  << 20)); // 48 MiB qkv bf16
  unsigned short* Qr  = (unsigned short*)(ws + (96u  << 20)); // 16 MiB
  unsigned short* Kr  = (unsigned short*)(ws + (112u << 20)); // 16 MiB
  unsigned short* Vt  = (unsigned short*)(ws + (128u << 20)); // 16 MiB
  unsigned short* ybf = (unsigned short*)(ws + (144u << 20)); // 16 MiB

  k_cvt_bf16<<<4096, 256, 0, stream>>>(x, xbf, MM * CC);
  k_transpose_bf16<<<8192, 256, 0, stream>>>(W_attn, WtA, CC, C3);
  k_transpose_bf16<<<4096, 256, 0, stream>>>(W_proj, WtP, CC, CC);

  // GEMM1: M=4096 N=6144 K=2048 -> (M/32)*(N/64) = 12288 waves -> 3072 blocks
  k_gemm_bf16<<<3072, 128, 0, stream>>>(xbf, WtA, b_attn, qkv, nullptr,
                                        MM, C3, CC, 0);
  k_rope<<<16384, 256, 0, stream>>>(qkv, Qr, Kr, Vt);

  // attention: B*H*(T/16) = 4096 waves -> 1024 blocks of 4 waves
  k_attn<<<1024, 128, 0, stream>>>(Qr, Kr, Vt, ybf);

  // GEMM2: M=4096 N=2048 K=2048 -> 4096 waves -> 1024 blocks
  k_gemm_bf16<<<1024, 128, 0, stream>>>(ybf, WtP, b_proj, nullptr, out,
                                        MM, CC, CC, 1);
}
